// ProtoSSMv5_30021821399780
// MI455X (gfx1250) — matrix operand order repaired and checked by reading, not compiled, not run on hardware
//
#include <hip/hip_runtime.h>
#include <math.h>

// ---------------- problem constants ----------------
#define B_    256
#define T_    12
#define DIN   1536
#define DM    320
#define DS    32
#define NL    4
#define NC    234
#define NH    8
#define HD    40      // DM / NH
#define KC    4
#define MD    24
#define ROWS  (B_ * T_)   // 3072

typedef __attribute__((ext_vector_type(16))) _Float16 v16h;
typedef __attribute__((ext_vector_type(8)))  _Float16 v8h;
typedef __attribute__((ext_vector_type(8)))  float    v8f;

// ---------------- wave32 reduction ----------------
__device__ __forceinline__ float wave_sum(float v) {
  #pragma unroll
  for (int off = 16; off > 0; off >>= 1) v += __shfl_xor(v, off, 32);
  return v;
}

__device__ __forceinline__ v8h cvt8(float4 x, float4 y) {
  v8h p;
  p[0] = (_Float16)x.x; p[1] = (_Float16)x.y;
  p[2] = (_Float16)x.z; p[3] = (_Float16)x.w;
  p[4] = (_Float16)y.x; p[5] = (_Float16)y.y;
  p[6] = (_Float16)y.z; p[7] = (_Float16)y.w;
  return p;
}

__device__ __forceinline__ float4 scale4(float4 x, float s) {
  return make_float4(x.x * s, x.y * s, x.z * s, x.w * s);
}

// =====================================================================
// Tiled WMMA GEMM:  C[M,N] = A[M,K] @ W[N,K]^T (+bias, +optional GELU)
// fp32 in/out, f16 compute / f32 accumulate (v_wmma_f32_16x16x32_f16).
//
// ASSUMPTIONS (hold for every launch in this pipeline):
//   M % 128 == 0, K % 32 == 0. Only N may be ragged (bounds on W rows
//   handled branchlessly by clamp+mask; C stores guarded on n < N).
//
// Block = 256 threads = 8 wave32 waves arranged 4x2; each wave owns a
// 32x32 C macro-tile = 2x2 WMMA tiles -> 4 v_wmma per K-step with A/B
// fragment reuse. Register double-buffering overlaps the next K-chunk's
// HBM loads with the current chunk's WMMAs.
// =====================================================================
#define TM 128
#define TN 64
#define TK 32

__global__ __launch_bounds__(256) void gemm_wmma(
    const float* __restrict__ A, int lda,
    const float* __restrict__ W, int ldw,   // W is [N, K] row-major == B^T
    const float* __restrict__ bias,         // length N or nullptr
    float* __restrict__ C, int ldc,
    int M, int N, int K, int act)           // act: 0=none, 1=exact GELU
{
  __shared__ __align__(32) _Float16 sA[TM][TK];  // 128x32 f16 = 8 KB
  __shared__ __align__(32) _Float16 sW[TN][TK];  //  64x32 f16 = 4 KB

  const int tid  = threadIdx.x;
  const int lane = tid & 31;
  const int wave = tid >> 5;
  const int wm = wave & 3;       // 0..3 row macro-tile
  const int wn = wave >> 2;      // 0..1 col macro-tile
  const int m0 = blockIdx.x * TM;
  const int n0 = blockIdx.y * TN;

  const int l16 = lane & 15;     // m (A-frag) / n (B-frag) within 16x16 tile
  const int kh  = lane >> 4;     // K-half selector per ISA 7.12.2

  // ---- staging coordinates (branchless; only W rows can be ragged) ----
  const int ar  = tid >> 1;          // 0..127  A row within tile
  const int as  = (tid & 1) * 16;    // 0 | 16  A k-offset (16 floats)
  const int wr  = tid >> 2;          // 0..63   W row within tile
  const int ws_ = (tid & 3) * 8;     // 0..24   W k-offset (8 floats)
  const int gn  = n0 + wr;
  const int cn  = gn < N ? gn : N - 1;
  const float wmask = gn < N ? 1.f : 0.f;

  const float* Ap = A + (size_t)(m0 + ar) * lda + as;   // + k0 each step
  const float* Wp = W + (size_t)cn * ldw + ws_;

  v8f acc[2][2] = {};

  // register-stage K-chunk 0
  float4 a0 = *(const float4*)(Ap + 0);
  float4 a1 = *(const float4*)(Ap + 4);
  float4 a2 = *(const float4*)(Ap + 8);
  float4 a3 = *(const float4*)(Ap + 12);
  float4 w0 = *(const float4*)(Wp + 0);
  float4 w1 = *(const float4*)(Wp + 4);

  for (int k0 = 0; k0 < K; k0 += TK) {
    // ---- commit staged registers to LDS (fp32 -> f16) ----
    *(v8h*)&sA[ar][as]     = cvt8(a0, a1);
    *(v8h*)&sA[ar][as + 8] = cvt8(a2, a3);
    *(v8h*)&sW[wr][ws_]    = cvt8(scale4(w0, wmask), scale4(w1, wmask));
    __syncthreads();

    // ---- issue next K-chunk's global loads (overlap with WMMAs) ----
    if (k0 + TK < K) {
      const float* An = Ap + k0 + TK;
      const float* Wn = Wp + k0 + TK;
      a0 = *(const float4*)(An + 0);
      a1 = *(const float4*)(An + 4);
      a2 = *(const float4*)(An + 8);
      a3 = *(const float4*)(An + 12);
      w0 = *(const float4*)(Wn + 0);
      w1 = *(const float4*)(Wn + 4);
    }
    // ---- prefetch the chunk after next (global_prefetch_b8) ----
    if (k0 + 2 * TK < K) {
      if (tid < TM) {
        __builtin_prefetch(A + (size_t)(m0 + tid) * lda + k0 + 2 * TK, 0, 1);
      } else if (tid < TM + TN) {
        int r = tid - TM;
        int cr = (n0 + r) < N ? (n0 + r) : N - 1;
        __builtin_prefetch(W + (size_t)cr * ldw + k0 + 2 * TK, 0, 1);
      }
    }

    // ---- fragment gather (ISA 7.12.2) + 4 WMMAs, 2x2 reuse ----
    v16h af[2], bf[2];
    #pragma unroll
    for (int i = 0; i < 2; ++i) {
      const _Float16* arow = &sA[wm * 32 + 16 * i + l16][0];
      v8h lo = *(const v8h*)(arow + 8 * kh);        // k = 8*kh .. +7
      v8h hi = *(const v8h*)(arow + 16 + 8 * kh);   // k = 16+8*kh .. +7
      #pragma unroll
      for (int j = 0; j < 8; ++j) { af[i][j] = lo[j]; af[i][8 + j] = hi[j]; }
      const _Float16* wrow = &sW[wn * 32 + 16 * i + l16][0];
      v8h blo = *(const v8h*)(wrow + 8 * kh);
      v8h bhi = *(const v8h*)(wrow + 16 + 8 * kh);
      #pragma unroll
      for (int j = 0; j < 8; ++j) { bf[i][j] = blo[j]; bf[i][8 + j] = bhi[j]; }
    }
    acc[0][0] = __builtin_amdgcn_wmma_f32_16x16x32_f16(false, af[0], false, bf[0], (short)0, acc[0][0], false, false);
    acc[0][1] = __builtin_amdgcn_wmma_f32_16x16x32_f16(false, af[0], false, bf[1], (short)0, acc[0][1], false, false);
    acc[1][0] = __builtin_amdgcn_wmma_f32_16x16x32_f16(false, af[1], false, bf[0], (short)0, acc[1][0], false, false);
    acc[1][1] = __builtin_amdgcn_wmma_f32_16x16x32_f16(false, af[1], false, bf[1], (short)0, acc[1][1], false, false);

    __syncthreads();
  }

  // ---- epilogue: C layout -> VGPR r: lanes0-15 M=r, lanes16-31 M=r+8 ----
  #pragma unroll
  for (int i = 0; i < 2; ++i) {
    #pragma unroll
    for (int j = 0; j < 2; ++j) {
      const int n = n0 + wn * 32 + 16 * j + l16;
      if (n < N) {
        #pragma unroll
        for (int r = 0; r < 8; ++r) {
          const int m = m0 + wm * 32 + 16 * i + r + 8 * kh;
          float v = acc[i][j][r];
          if (bias) v += bias[n];
          if (act == 1) v = 0.5f * v * (1.f + erff(v * 0.70710678118f));
          C[(size_t)m * ldc + n] = v;
        }
      }
    }
  }
}

// =====================================================================
// meta embedding projection: out[b,d] = meta_W[d,:] . [site_emb|hour_emb]
// =====================================================================
__global__ void metaproj_kernel(const float* __restrict__ site_emb,
                                const float* __restrict__ hour_emb,
                                const int* __restrict__ site_ids,
                                const int* __restrict__ hours,
                                const float* __restrict__ Wm,
                                const float* __restrict__ bm,
                                float* __restrict__ out) {
  int i = blockIdx.x * blockDim.x + threadIdx.x;
  if (i >= B_ * DM) return;
  int b = i / DM, d = i % DM;
  int s = site_ids[b], hr = hours[b];
  const float* wrow = Wm + (size_t)d * (2 * MD);
  float acc = bm[d];
  #pragma unroll
  for (int j = 0; j < MD; ++j) acc += wrow[j]      * site_emb[s  * MD + j];
  #pragma unroll
  for (int j = 0; j < MD; ++j) acc += wrow[MD + j] * hour_emb[hr * MD + j];
  out[i] = acc;
}

// =====================================================================
// post input-projection: h = GELU(LN(tmp)) + pos[t] + metaproj[b]
// one wave32 per row, D=320 -> 10 elems/lane
// =====================================================================
__global__ void embed_post_kernel(const float* __restrict__ tmp,
                                  const float* __restrict__ g_,
                                  const float* __restrict__ b_,
                                  const float* __restrict__ pos,
                                  const float* __restrict__ mp,
                                  float* __restrict__ h) {
  int row  = blockIdx.x * (blockDim.x >> 5) + (threadIdx.x >> 5);
  int lane = threadIdx.x & 31;
  if (row >= ROWS) return;
  int b = row / T_, t = row % T_;
  const float* x = tmp + (size_t)row * DM;
  float v[10]; float s = 0.f, s2 = 0.f;
  #pragma unroll
  for (int i = 0; i < 10; ++i) {
    v[i] = x[lane + 32 * i]; s += v[i]; s2 += v[i] * v[i];
  }
  s = wave_sum(s); s2 = wave_sum(s2);
  float mean = s / DM;
  float rstd = rsqrtf(s2 / DM - mean * mean + 1e-5f);
  float* out = h + (size_t)row * DM;
  #pragma unroll
  for (int i = 0; i < 10; ++i) {
    int d = lane + 32 * i;
    float xn = (v[i] - mean) * rstd * g_[d] + b_[d];
    float ge = 0.5f * xn * (1.f + erff(xn * 0.70710678118f));
    out[d] = ge + pos[t * DM + d] + mp[b * DM + d];
  }
}

// =====================================================================
// residual + LayerNorm in place:  h = LN(h + delta) * g + b
// =====================================================================
__global__ void addln_kernel(float* __restrict__ h,
                             const float* __restrict__ delta,
                             const float* __restrict__ g_,
                             const float* __restrict__ b_) {
  int row  = blockIdx.x * (blockDim.x >> 5) + (threadIdx.x >> 5);
  int lane = threadIdx.x & 31;
  if (row >= ROWS) return;
  float* x = h + (size_t)row * DM;
  const float* dd = delta + (size_t)row * DM;
  float v[10]; float s = 0.f, s2 = 0.f;
  #pragma unroll
  for (int i = 0; i < 10; ++i) {
    v[i] = x[lane + 32 * i] + dd[lane + 32 * i];
    s += v[i]; s2 += v[i] * v[i];
  }
  s = wave_sum(s); s2 = wave_sum(s2);
  float mean = s / DM;
  float rstd = rsqrtf(s2 / DM - mean * mean + 1e-5f);
  #pragma unroll
  for (int i = 0; i < 10; ++i) {
    int d = lane + 32 * i;
    x[d] = (v[i] - mean) * rstd * g_[d] + b_[d];
  }
}

// =====================================================================
// causal (dir=0) / anti-causal (dir=1, == flip-conv-flip) depthwise conv
// =====================================================================
__global__ void conv_kernel(const float* __restrict__ xz, int ld,
                            const float* __restrict__ cw,
                            const float* __restrict__ cb,
                            float* __restrict__ xconv, int dir) {
  int i = blockIdx.x * blockDim.x + threadIdx.x;
  if (i >= B_ * DM) return;
  int b = i / DM, d = i % DM;
  const float* w = cw + d * KC;
  for (int t = 0; t < T_; ++t) {
    float acc = cb[d];
    #pragma unroll
    for (int k = 0; k < KC; ++k) {
      int tt = dir ? (t + (KC - 1) - k) : (t - (KC - 1) + k);
      if (tt >= 0 && tt < T_) acc += w[k] * xz[(size_t)(b * T_ + tt) * ld + d];
    }
    xconv[(size_t)(b * T_ + t) * DM + d] = acc;
  }
}

// =====================================================================
// selective scan: one thread per (b,d), 32-float state in VGPRs.
// dir flag walks time forward/backward (== flip-scan-flip).
// softplus fused on dt; y = C.h + Dp*x
// =====================================================================
__global__ void scan_kernel(const float* __restrict__ xconv,
                            const float* __restrict__ dtpre,
                            const float* __restrict__ Bm,
                            const float* __restrict__ Cm,
                            const float* __restrict__ A_log,
                            const float* __restrict__ Dp,
                            float* __restrict__ y, int dir) {
  int i = blockIdx.x * blockDim.x + threadIdx.x;
  if (i >= B_ * DM) return;
  int b = i / DM, d = i % DM;
  float a[DS], st[DS];
  #pragma unroll
  for (int s = 0; s < DS; ++s) { a[s] = -expf(A_log[d * DS + s]); st[s] = 0.f; }
  float dp = Dp[d];
  for (int ii = 0; ii < T_; ++ii) {
    int t = dir ? (T_ - 1 - ii) : ii;
    size_t rid = (size_t)(b * T_ + t);
    float dtr = dtpre[rid * DM + d];
    float dt  = dtr > 20.f ? dtr : log1pf(expf(dtr));
    float xv  = xconv[rid * DM + d];
    const float* Bt = Bm + rid * DS;
    const float* Ct = Cm + rid * DS;
    float dx = dt * xv, acc = 0.f;
    #pragma unroll
    for (int s = 0; s < DS; ++s) {
      st[s] = st[s] * expf(dt * a[s]) + dx * Bt[s];
      acc += st[s] * Ct[s];
    }
    y[rid * DM + d] = acc + xv * dp;
  }
}

// g = y * silu(z), z = xz[:, DM:2*DM]
__global__ void gate_kernel(const float* __restrict__ y,
                            const float* __restrict__ xz, int ld,
                            float* __restrict__ g) {
  int i = blockIdx.x * blockDim.x + threadIdx.x;
  if (i >= ROWS * DM) return;
  int row = i / DM, d = i % DM;
  float z = xz[(size_t)row * ld + DM + d];
  g[i] = y[i] * (z / (1.f + expf(-z)));
}

// =====================================================================
// tiny attention (T=12): one thread per (b, head, q)
// =====================================================================
__global__ void attn_kernel(const float* __restrict__ qkv, float* __restrict__ out) {
  int i = blockIdx.x * blockDim.x + threadIdx.x;
  if (i >= B_ * NH * T_) return;
  int q  = i % T_;
  int bh = i / T_;
  int hh = bh % NH, b = bh / NH;
  const float* qv = qkv + (size_t)(b * T_ + q) * (3 * DM) + hh * HD;
  float sc[T_], mx = -1e30f;
  for (int k = 0; k < T_; ++k) {
    const float* kv = qkv + (size_t)(b * T_ + k) * (3 * DM) + DM + hh * HD;
    float acc = 0.f;
    #pragma unroll
    for (int d = 0; d < HD; ++d) acc += qv[d] * kv[d];
    sc[k] = acc * 0.158113883f; // 1/sqrt(40)
    mx = fmaxf(mx, sc[k]);
  }
  float den = 0.f;
  for (int k = 0; k < T_; ++k) { sc[k] = expf(sc[k] - mx); den += sc[k]; }
  float inv = 1.f / den;
  float o[HD];
  #pragma unroll
  for (int d = 0; d < HD; ++d) o[d] = 0.f;
  for (int k = 0; k < T_; ++k) {
    const float* vv = qkv + (size_t)(b * T_ + k) * (3 * DM) + 2 * DM + hh * HD;
    float w = sc[k] * inv;
    #pragma unroll
    for (int d = 0; d < HD; ++d) o[d] += w * vv[d];
  }
  float* op = out + (size_t)(b * T_ + q) * DM + hh * HD;
  #pragma unroll
  for (int d = 0; d < HD; ++d) op[d] = o[d];
}

// row-normalize h -> hn, and copy raw h into the output's h segment
__global__ void rownorm_kernel(const float* __restrict__ h,
                               float* __restrict__ hn,
                               float* __restrict__ hout) {
  int row  = blockIdx.x * (blockDim.x >> 5) + (threadIdx.x >> 5);
  int lane = threadIdx.x & 31;
  if (row >= ROWS) return;
  const float* x = h + (size_t)row * DM;
  float v[10]; float s2 = 0.f;
  #pragma unroll
  for (int i = 0; i < 10; ++i) { v[i] = x[lane + 32 * i]; s2 += v[i] * v[i]; }
  s2 = wave_sum(s2);
  float inv = 1.f / fmaxf(sqrtf(s2), 1e-12f);
  #pragma unroll
  for (int i = 0; i < 10; ++i) {
    int d = lane + 32 * i;
    hn[(size_t)row * DM + d]   = v[i] * inv;
    hout[(size_t)row * DM + d] = v[i];
  }
}

// prototype L2-normalization (234 rows x 320)
__global__ void protonorm_kernel(const float* __restrict__ protos,
                                 float* __restrict__ pn) {
  int row  = blockIdx.x * (blockDim.x >> 5) + (threadIdx.x >> 5);
  int lane = threadIdx.x & 31;
  if (row >= NC) return;
  const float* x = protos + (size_t)row * DM;
  float v[10]; float s2 = 0.f;
  #pragma unroll
  for (int i = 0; i < 10; ++i) { v[i] = x[lane + 32 * i]; s2 += v[i] * v[i]; }
  s2 = wave_sum(s2);
  float inv = 1.f / fmaxf(sqrtf(s2), 1e-12f);
  #pragma unroll
  for (int i = 0; i < 10; ++i)
    pn[(size_t)row * DM + lane + 32 * i] = v[i] * inv;
}

// species_logits = sigmoid(alpha)*sim*clip(temp) + (1-sigmoid(alpha))*perch
__global__ void logits_kernel(const float* __restrict__ sim,
                              const float* __restrict__ temp,
                              const float* __restrict__ alpha,
                              const float* __restrict__ perch,
                              float* __restrict__ out) {
  int i = blockIdx.x * blockDim.x + threadIdx.x;
  if (i >= ROWS * NC) return;
  int c = i % NC;
  float tt = fminf(fmaxf(temp[c], 0.1f), 10.f);
  float al = 1.f / (1.f + expf(-alpha[c]));
  out[i] = al * sim[i] * tt + (1.f - al) * perch[i];
}

// =====================================================================
// host orchestration
// =====================================================================
struct SSMP {
  const float *A_log, *Dp, *W_B, *W_C, *W_dt, *W_in, *W_out, *b_dt, *conv_b, *conv_w;
};

extern "C" void kernel_launch(void* const* d_in, const int* in_sizes, int n_in,
                              void* d_out, int out_size, void* d_ws, size_t ws_size,
                              hipStream_t stream) {
  (void)in_sizes; (void)n_in; (void)out_size; (void)ws_size;

  // --- input leaves in jax sorted-dict flatten order ---
  const float* emb      = (const float*)d_in[0];
  const float* perch    = (const float*)d_in[1];
  const int*   site_ids = (const int*)  d_in[2];
  const int*   hours    = (const int*)  d_in[3];
  const float* alpha    = (const float*)d_in[4];
  const float* Wo       = (const float*)d_in[5];
  const float* Wqkv     = (const float*)d_in[6];
  const float* bo       = (const float*)d_in[7];
  const float* bqkv     = (const float*)d_in[8];
  const float* f1W      = (const float*)d_in[9];
  const float* f1b      = (const float*)d_in[10];
  const float* f2W      = (const float*)d_in[11];
  const float* f2b      = (const float*)d_in[12];
  const float* n1b      = (const float*)d_in[13];
  const float* n1g      = (const float*)d_in[14];
  const float* n2b      = (const float*)d_in[15];
  const float* n2g      = (const float*)d_in[16];
  const float* hour_emb = (const float*)d_in[17];
  const float* in_W     = (const float*)d_in[18];
  const float* in_b     = (const float*)d_in[19];
  const float* in_beta  = (const float*)d_in[20];
  const float* in_g     = (const float*)d_in[21];
  const float* meta_W   = (const float*)d_in[118];
  const float* meta_b   = (const float*)d_in[119];
  const float* pos      = (const float*)d_in[120];
  const float* protos   = (const float*)d_in[121];
  const float* site_emb = (const float*)d_in[122];
  const float* temp     = (const float*)d_in[123];

  auto get_ssm = [&](int base) {
    SSMP p;
    p.A_log  = (const float*)d_in[base + 0];
    p.Dp     = (const float*)d_in[base + 1];
    p.W_B    = (const float*)d_in[base + 2];
    p.W_C    = (const float*)d_in[base + 3];
    p.W_dt   = (const float*)d_in[base + 4];
    p.W_in   = (const float*)d_in[base + 5];
    p.W_out  = (const float*)d_in[base + 6];
    p.b_dt   = (const float*)d_in[base + 7];
    p.conv_b = (const float*)d_in[base + 8];
    p.conv_w = (const float*)d_in[base + 9];
    return p;
  };

  // --- workspace arena (floats) ---
  float* W0 = (float*)d_ws;
  size_t o = 0;
  float* h       = W0 + o; o += (size_t)ROWS * DM;        // residual stream
  float* bufXL   = W0 + o; o += (size_t)ROWS * 3 * DM;    // xz(ld640) / qkv(ld960) / ff1(ld640)
  float* xconv   = W0 + o; o += (size_t)ROWS * DM;
  float* dtpre   = W0 + o; o += (size_t)ROWS * DM;
  float* Bm      = W0 + o; o += (size_t)ROWS * DS;
  float* Cm      = W0 + o; o += (size_t)ROWS * DS;
  float* ybuf    = W0 + o; o += (size_t)ROWS * DM;
  float* gbuf    = W0 + o; o += (size_t)ROWS * DM;        // gate out / attn out
  float* cat     = W0 + o; o += (size_t)ROWS * 2 * DM;    // [h_f | h_b]
  float* tmp320  = W0 + o; o += (size_t)ROWS * DM;        // in-proj / hm / proj / ff2 / sim
  float* mproj   = W0 + o; o += (size_t)B_ * DM;
  float* pn      = W0 + o; o += (size_t)NC * DM;
  float* hn      = W0 + o; o += (size_t)ROWS * DM;

  float* out_logits = (float*)d_out;                      // (B,T,NC)
  float* out_h      = out_logits + (size_t)ROWS * NC;     // (B,T,DM)

  auto GEMM = [&](const float* A, int lda, const float* Wt, int ldw,
                  const float* bias, float* C, int ldc,
                  int M, int N, int K, int act) {
    dim3 grid((M + TM - 1) / TM, (N + TN - 1) / TN);
    gemm_wmma<<<grid, dim3(256), 0, stream>>>(A, lda, Wt, ldw, bias, C, ldc, M, N, K, act);
  };

  const int EW = 256;
  const int rows_grid = ROWS / 8;               // wave-per-row kernels, 8 waves/block

  // ---- stem: meta proj, input GEMM, LN+GELU+pos+meta ----
  metaproj_kernel<<<(B_ * DM + EW - 1) / EW, EW, 0, stream>>>(
      site_emb, hour_emb, site_ids, hours, meta_W, meta_b, mproj);
  GEMM(emb, DIN, in_W, DIN, in_b, tmp320, DM, ROWS, DM, DIN, 0);
  embed_post_kernel<<<rows_grid, 256, 0, stream>>>(tmp320, in_g, in_beta, pos, mproj, h);

  // ---- bidirectional Mamba layers ----
  for (int l = 0; l < NL; ++l) {
    int base = 22 + 24 * l;
    SSMP bwd = get_ssm(base + 0);
    SSMP fwd = get_ssm(base + 10);
    const float* mW = (const float*)d_in[base + 20];
    const float* mb = (const float*)d_in[base + 21];
    const float* nb = (const float*)d_in[base + 22];
    const float* ng = (const float*)d_in[base + 23];

    for (int dir = 0; dir < 2; ++dir) {
      const SSMP& p = dir ? bwd : fwd;
      // xz = h @ W_in^T  (ROWS x 640)
      GEMM(h, DM, p.W_in, DM, nullptr, bufXL, 2 * DM, ROWS, 2 * DM, DM, 0);
      conv_kernel<<<(B_ * DM + EW - 1) / EW, EW, 0, stream>>>(
          bufXL, 2 * DM, p.conv_w, p.conv_b, xconv, dir);
      GEMM(xconv, DM, p.W_dt, DM, p.b_dt, dtpre, DM, ROWS, DM, DM, 0);
      GEMM(xconv, DM, p.W_B,  DM, nullptr, Bm, DS, ROWS, DS, DM, 0);
      GEMM(xconv, DM, p.W_C,  DM, nullptr, Cm, DS, ROWS, DS, DM, 0);
      scan_kernel<<<(B_ * DM + EW - 1) / EW, EW, 0, stream>>>(
          xconv, dtpre, Bm, Cm, p.A_log, p.Dp, ybuf, dir);
      gate_kernel<<<(ROWS * DM + EW - 1) / EW, EW, 0, stream>>>(
          ybuf, bufXL, 2 * DM, gbuf);
      // h_dir = g @ W_out^T, written into its half of the concat buffer
      GEMM(gbuf, DM, p.W_out, DM, nullptr, cat + dir * DM, 2 * DM, ROWS, DM, DM, 0);
    }
    // hm = cat @ mW^T + mb ; h = LN(h + hm)
    GEMM(cat, 2 * DM, mW, 2 * DM, mb, tmp320, DM, ROWS, DM, 2 * DM, 0);
    addln_kernel<<<rows_grid, 256, 0, stream>>>(h, tmp320, ng, nb);
  }

  // ---- attention block ----
  GEMM(h, DM, Wqkv, DM, bqkv, bufXL, 3 * DM, ROWS, 3 * DM, DM, 0);
  attn_kernel<<<(B_ * NH * T_ + EW - 1) / EW, EW, 0, stream>>>(bufXL, gbuf);
  GEMM(gbuf, DM, Wo, DM, bo, tmp320, DM, ROWS, DM, DM, 0);
  addln_kernel<<<rows_grid, 256, 0, stream>>>(h, tmp320, n1g, n1b);

  // ---- FFN (GELU fused in GEMM epilogue) ----
  GEMM(h, DM, f1W, DM, f1b, bufXL, 2 * DM, ROWS, 2 * DM, DM, 1);
  GEMM(bufXL, 2 * DM, f2W, 2 * DM, f2b, tmp320, DM, ROWS, DM, 2 * DM, 0);
  addln_kernel<<<rows_grid, 256, 0, stream>>>(h, tmp320, n2g, n2b);

  // ---- prototype head ----
  protonorm_kernel<<<(NC + 7) / 8, 256, 0, stream>>>(protos, pn);
  rownorm_kernel<<<rows_grid, 256, 0, stream>>>(h, hn, out_h);
  GEMM(hn, DM, pn, DM, nullptr, tmp320, NC, ROWS, NC, DM, 0);   // sim (ld = NC)
  logits_kernel<<<(ROWS * NC + EW - 1) / EW, EW, 0, stream>>>(
      tmp320, temp, alpha, perch, out_logits);
}
